// Mamba1_83932250898432
// MI455X (gfx1250) — matrix operand order repaired and checked by reading, not compiled, not run on hardware
//
#include <hip/hip_runtime.h>
#include <hip/hip_bf16.h>
#include <cstddef>
#include <cstdint>

#define B_SZ   2
#define L_SZ   2048
#define DMODEL 1024
#define DINNER 2048
#define DSTATE 16
#define DCONV  4
#define DTRANK 64
#define MTOT   (B_SZ * L_SZ)         // 4096 rows for all GEMMs
#define NXZ    (2 * DINNER)          // 4096
#define NDBL   (DTRANK + 2 * DSTATE) // 96

typedef __attribute__((ext_vector_type(16))) _Float16 v16h;
typedef __attribute__((ext_vector_type(8)))  _Float16 v8h;
typedef __attribute__((ext_vector_type(8)))  float    v8f;

// ---------------------------------------------------------------------------
// f32 -> f16 convert
// ---------------------------------------------------------------------------
__global__ void cvt_f16_kernel(const float* __restrict__ src,
                               _Float16* __restrict__ dst, size_t n) {
    size_t i = (size_t)blockIdx.x * blockDim.x + threadIdx.x;
    if (i < n) dst[i] = (_Float16)src[i];
}

// ---------------------------------------------------------------------------
// WMMA GEMM:  C[M x N] = A[M x K] * B[N x K]^T   (A,B f16 row-major, C f32)
// (16*TM) x (16*TN) block per wave; LDA/LDB are compile-time so per-tile
// addresses are base + immediate offset (no pointer arrays -> no spills).
// k-loop unrolled x2 with alternating fragment sets (no buffer copies).
// Fragment layouts per CDNA5 ISA.
// ---------------------------------------------------------------------------
template <int TM, int TN, int LDA, int LDB>
__global__ void gemm_f16_wmma(const _Float16* __restrict__ A,
                              const _Float16* __restrict__ Bm,
                              float* __restrict__ C, int ldc,
                              int M, int N, int K) {
    const int lane  = threadIdx.x & 31;
    const int gwave = blockIdx.x * (blockDim.x >> 5) + (threadIdx.x >> 5);
    const int ngr   = N / (16 * TN);
    const int tm0   = (gwave / ngr) * (16 * TM);
    const int tn0   = (gwave % ngr) * (16 * TN);
    if (tm0 >= M) return;

    const int half = lane >> 4;   // 0: lanes 0-15, 1: lanes 16-31
    const int l15  = lane & 15;

    const _Float16* aB = A  + (size_t)(tm0 + l15) * LDA + half * 8;
    const _Float16* bB = Bm + (size_t)(tn0 + l15) * LDB + half * 8;

    v8f acc[TM][TN];
#pragma unroll
    for (int i = 0; i < TM; ++i)
#pragma unroll
        for (int j = 0; j < TN; ++j)
#pragma unroll
            for (int e = 0; e < 8; ++e) acc[i][j][e] = 0.0f;

    // fragment loads at compile-time tile offsets from the running base
    auto ldA = [](const _Float16* p, int i) -> v16h {
        v8h a0 = *(const v8h*)(p + (size_t)i * 16 * LDA);
        v8h a1 = *(const v8h*)(p + (size_t)i * 16 * LDA + 16);
        return __builtin_shufflevector(a0, a1, 0, 1, 2, 3, 4, 5, 6, 7,
                                       8, 9, 10, 11, 12, 13, 14, 15);
    };
    auto ldB = [](const _Float16* p, int j) -> v16h {
        v8h b0 = *(const v8h*)(p + (size_t)j * 16 * LDB);
        v8h b1 = *(const v8h*)(p + (size_t)j * 16 * LDB + 16);
        return __builtin_shufflevector(b0, b1, 0, 1, 2, 3, 4, 5, 6, 7,
                                       8, 9, 10, 11, 12, 13, 14, 15);
    };

    v16h a0[TM], b0[TN], a1[TM], b1[TN];

    // prologue: stage 0 holds k-step 0
#pragma unroll
    for (int i = 0; i < TM; ++i) a0[i] = ldA(aB, i);
#pragma unroll
    for (int j = 0; j < TN; ++j) b0[j] = ldB(bB, j);

    const int kIter = K / 32;   // even for all GEMMs here
    for (int it = 0; it < kIter - 2; it += 2) {
        aB += 32; bB += 32;
        __builtin_prefetch(aB + 128, 0, 1);
#pragma unroll
        for (int i = 0; i < TM; ++i) a1[i] = ldA(aB, i);
#pragma unroll
        for (int j = 0; j < TN; ++j) b1[j] = ldB(bB, j);
#pragma unroll
        for (int i = 0; i < TM; ++i)
#pragma unroll
            for (int j = 0; j < TN; ++j)
                acc[i][j] = __builtin_amdgcn_wmma_f32_16x16x32_f16(
                    false, a0[i], false, b0[j], (short)0, acc[i][j],
                    false, false);

        aB += 32; bB += 32;
#pragma unroll
        for (int i = 0; i < TM; ++i) a0[i] = ldA(aB, i);
#pragma unroll
        for (int j = 0; j < TN; ++j) b0[j] = ldB(bB, j);
#pragma unroll
        for (int i = 0; i < TM; ++i)
#pragma unroll
            for (int j = 0; j < TN; ++j)
                acc[i][j] = __builtin_amdgcn_wmma_f32_16x16x32_f16(
                    false, a1[i], false, b1[j], (short)0, acc[i][j],
                    false, false);
    }

    // tail: stage 0 holds k-step (kIter-2); load last step into stage 1
    aB += 32; bB += 32;
#pragma unroll
    for (int i = 0; i < TM; ++i) a1[i] = ldA(aB, i);
#pragma unroll
    for (int j = 0; j < TN; ++j) b1[j] = ldB(bB, j);
#pragma unroll
    for (int i = 0; i < TM; ++i)
#pragma unroll
        for (int j = 0; j < TN; ++j)
            acc[i][j] = __builtin_amdgcn_wmma_f32_16x16x32_f16(
                false, a0[i], false, b0[j], (short)0, acc[i][j], false, false);
#pragma unroll
    for (int i = 0; i < TM; ++i)
#pragma unroll
        for (int j = 0; j < TN; ++j)
            acc[i][j] = __builtin_amdgcn_wmma_f32_16x16x32_f16(
                false, a1[i], false, b1[j], (short)0, acc[i][j], false, false);

#pragma unroll
    for (int i = 0; i < TM; ++i)
#pragma unroll
        for (int j = 0; j < TN; ++j) {
            int col = tn0 + j * 16 + l15;
#pragma unroll
            for (int e = 0; e < 8; ++e) {
                int row = tm0 + i * 16 + e + half * 8;
                C[(size_t)row * ldc + col] = acc[i][j][e];
            }
        }
}

// ---------------------------------------------------------------------------
// Depthwise causal conv (k=4) + bias + SiLU on xi = xz[:, 0:DINNER]
// ---------------------------------------------------------------------------
__global__ void conv_silu_kernel(const float* __restrict__ xz,
                                 const float* __restrict__ conv_w,
                                 const float* __restrict__ conv_b,
                                 float* __restrict__ xc32,
                                 _Float16* __restrict__ xc16) {
    size_t idx = (size_t)blockIdx.x * blockDim.x + threadIdx.x;
    if (idx >= (size_t)MTOT * DINNER) return;
    int c  = (int)(idx % DINNER);
    int ml = (int)(idx / DINNER);
    int l  = ml % L_SZ;
    int b  = ml / L_SZ;
    float acc = conv_b[c];
#pragma unroll
    for (int j = 0; j < DCONV; ++j) {
        int ll = l - (DCONV - 1) + j;
        if (ll >= 0)
            acc += conv_w[c * DCONV + j] *
                   xz[((size_t)(b * L_SZ + ll)) * NXZ + c];
    }
    float s = acc / (1.0f + __expf(-acc));   // SiLU
    xc32[idx] = s;
    xc16[idx] = (_Float16)s;
}

// ---------------------------------------------------------------------------
// dt = softplus(dt_lin + dt_proj_b), in place on (MTOT x DINNER)
// ---------------------------------------------------------------------------
__global__ void dt_softplus_kernel(float* __restrict__ dt,
                                   const float* __restrict__ bias, size_t n) {
    size_t i = (size_t)blockIdx.x * blockDim.x + threadIdx.x;
    if (i >= n) return;
    float v = dt[i] + bias[i % DINNER];
    dt[i] = (v > 20.0f) ? v : log1pf(__expf(v));
}

// ---------------------------------------------------------------------------
// Selective scan: one lane per (b, channel). 16 states in registers.
// Fuses +xc*D, *silu(z), writes y as f16 for the out_proj GEMM.
// ---------------------------------------------------------------------------
#define SCAN_CHUNK 64
__global__ void scan_kernel(const float* __restrict__ dt,
                            const float* __restrict__ xc32,
                            const float* __restrict__ dbl,
                            const float* __restrict__ xz,
                            const float* __restrict__ A_log,
                            const float* __restrict__ Dp,
                            _Float16* __restrict__ y16) {
    const int b = blockIdx.x >> 3;                       // 8 blocks / batch
    const int c = (blockIdx.x & 7) * 256 + threadIdx.x;  // channel

    float Arow[DSTATE], h[DSTATE];
#pragma unroll
    for (int n = 0; n < DSTATE; ++n) {
        Arow[n] = -__expf(A_log[c * DSTATE + n]);
        h[n] = 0.0f;
    }
    const float Dv = Dp[c];

    __shared__ float sB[SCAN_CHUNK][DSTATE];
    __shared__ float sC[SCAN_CHUNK][DSTATE];

    for (int l0 = 0; l0 < L_SZ; l0 += SCAN_CHUNK) {
        __syncthreads();
        for (int t = threadIdx.x; t < SCAN_CHUNK * 32; t += 256) {
            int i = t >> 5;
            int w = t & 31;
            float v = dbl[(size_t)(b * L_SZ + l0 + i) * NDBL + DTRANK + w];
            if (w < DSTATE) sB[i][w] = v;
            else            sC[i][w - DSTATE] = v;
        }
        __syncthreads();

        for (int i = 0; i < SCAN_CHUNK; ++i) {
            size_t m  = (size_t)(b * L_SZ + l0 + i);
            float dtv = dt[m * DINNER + c];
            float xcv = xc32[m * DINNER + c];
            float dx  = dtv * xcv;
            float y   = 0.0f;
#pragma unroll
            for (int n = 0; n < DSTATE; ++n) {
                float ab = 2.0f * __expf(dtv * Arow[n]) - 1.0f;  // Abar
                h[n] = ab * h[n] + dx * sB[i][n];
                y += h[n] * sC[i][n];
            }
            y += xcv * Dv;
            float zv = xz[m * NXZ + DINNER + c];
            y *= zv / (1.0f + __expf(-zv));                      // * silu(z)
            y16[m * DINNER + c] = (_Float16)y;
        }
    }
}

// ---------------------------------------------------------------------------
// Host-side orchestration
// ---------------------------------------------------------------------------
extern "C" void kernel_launch(void* const* d_in, const int* in_sizes, int n_in,
                              void* d_out, int out_size, void* d_ws,
                              size_t ws_size, hipStream_t stream) {
    (void)in_sizes; (void)n_in; (void)out_size; (void)ws_size;

    const float* x          = (const float*)d_in[0];
    const float* in_proj_w  = (const float*)d_in[1];
    const float* conv_w     = (const float*)d_in[2];
    const float* conv_b     = (const float*)d_in[3];
    const float* x_proj_w   = (const float*)d_in[4];
    const float* dt_proj_w  = (const float*)d_in[5];
    const float* dt_proj_b  = (const float*)d_in[6];
    const float* A_log      = (const float*)d_in[7];
    const float* Dp         = (const float*)d_in[8];
    const float* out_proj_w = (const float*)d_in[9];
    float* out = (float*)d_out;

    char* ws = (char*)d_ws;
    size_t off = 0;
    auto alloc = [&](size_t bytes) -> void* {
        void* p = ws + off;
        off += (bytes + 255) & ~(size_t)255;
        return p;
    };

    _Float16* x16   = (_Float16*)alloc((size_t)MTOT * DMODEL * 2);
    _Float16* wi16  = (_Float16*)alloc((size_t)NXZ * DMODEL * 2);
    _Float16* wx16  = (_Float16*)alloc((size_t)NDBL * DINNER * 2);
    _Float16* wd16  = (_Float16*)alloc((size_t)DINNER * DTRANK * 2);
    _Float16* wo16  = (_Float16*)alloc((size_t)DMODEL * DINNER * 2);
    float*    xz    = (float*)   alloc((size_t)MTOT * NXZ * 4);
    float*    xc32  = (float*)   alloc((size_t)MTOT * DINNER * 4);
    _Float16* xc16  = (_Float16*)alloc((size_t)MTOT * DINNER * 2);
    float*    dbl   = (float*)   alloc((size_t)MTOT * NDBL * 4);
    _Float16* dbl16 = (_Float16*)alloc((size_t)MTOT * NDBL * 2);
    float*    dt32  = (float*)   alloc((size_t)MTOT * DINNER * 4);
    _Float16* y16   = (_Float16*)alloc((size_t)MTOT * DINNER * 2);

    auto cvt = [&](const float* s, _Float16* d, size_t n) {
        cvt_f16_kernel<<<(unsigned)((n + 255) / 256), 256, 0, stream>>>(s, d, n);
    };
    cvt(x,          x16,  (size_t)MTOT * DMODEL);
    cvt(in_proj_w,  wi16, (size_t)NXZ * DMODEL);
    cvt(x_proj_w,   wx16, (size_t)NDBL * DINNER);
    cvt(dt_proj_w,  wd16, (size_t)DINNER * DTRANK);
    cvt(out_proj_w, wo16, (size_t)DMODEL * DINNER);

    // 1) xz = x @ in_proj_w^T   (4096 x 4096, K=1024); 32x64 tile/wave
    {
        int waves = (MTOT / 32) * (NXZ / 64);   // 8192
        gemm_f16_wmma<2, 4, DMODEL, DMODEL><<<waves / 8, 256, 0, stream>>>(
            x16, wi16, xz, NXZ, MTOT, NXZ, DMODEL);
    }

    // 2) depthwise causal conv + SiLU -> xc (f32 + f16)
    {
        size_t n = (size_t)MTOT * DINNER;
        conv_silu_kernel<<<(unsigned)((n + 255) / 256), 256, 0, stream>>>(
            xz, conv_w, conv_b, xc32, xc16);
    }

    // 3) dbl = xc @ x_proj_w^T   (4096 x 96, K=2048); 32x48 tile/wave
    {
        int waves = (MTOT / 32) * (NDBL / 48);  // 256
        gemm_f16_wmma<2, 3, DINNER, DINNER><<<waves / 8, 256, 0, stream>>>(
            xc16, wx16, dbl, NDBL, MTOT, NDBL, DINNER);
    }
    cvt(dbl, dbl16, (size_t)MTOT * NDBL);

    // 4) dt_lin = dt_r @ dt_proj_w^T  (A = first 64 cols of dbl16, lda=96)
    {
        int waves = (MTOT / 32) * (DINNER / 64); // 4096
        gemm_f16_wmma<2, 4, NDBL, DTRANK><<<waves / 8, 256, 0, stream>>>(
            dbl16, wd16, dt32, DINNER, MTOT, DINNER, DTRANK);
    }

    // 5) dt = softplus(dt_lin + bias), in place
    {
        size_t n = (size_t)MTOT * DINNER;
        dt_softplus_kernel<<<(unsigned)((n + 255) / 256), 256, 0, stream>>>(
            dt32, dt_proj_b, n);
    }

    // 6) selective scan, fused D-skip + silu(z) gating -> y16
    scan_kernel<<<B_SZ * (DINNER / 256), 256, 0, stream>>>(
        dt32, xc32, dbl, xz, A_log, Dp, y16);

    // 7) out = y @ out_proj_w^T  (4096 x 1024, K=2048); 32x64 tile/wave
    {
        int waves = (MTOT / 32) * (DMODEL / 64); // 2048
        gemm_f16_wmma<2, 4, DINNER, DINNER><<<waves / 8, 256, 0, stream>>>(
            y16, wo16, out, DMODEL, MTOT, DMODEL, DINNER);
    }
}
